// MultiHeadAttn_4389456577008
// MI455X (gfx1250) — matrix operand order repaired and checked by reading, not compiled, not run on hardware
//
#include <hip/hip_runtime.h>

#define B_  2
#define S_  2048
#define D_  1024
#define H_  16
#define HD_ 64
#define EPS_ 1e-6f

typedef __attribute__((ext_vector_type(16))) __bf16 v16bf;
typedef __attribute__((ext_vector_type(8)))  float  v8f;

union FragU { v16bf v; uint4 u[2]; };
union PackU { uint4 u; unsigned short s[8]; };

// async direct global->LDS copy, 16 bytes, tracked by ASYNCcnt (CDNA5 §15.18.3 op 98)
#define ASYNC_G2L_B128(ldsaddr_u32, gptr) \
  asm volatile("global_load_async_to_lds_b128 %0, %1, off" \
               :: "v"(ldsaddr_u32), "v"(gptr) : "memory")

__device__ __forceinline__ unsigned short f2bf_bits(float x) {
  union { float f; unsigned u; } a; a.f = x;
  unsigned r = a.u + 0x7FFFu + ((a.u >> 16) & 1u);   // round-to-nearest-even
  return (unsigned short)(r >> 16);
}

// A-fragment (16xK=32, bf16): lanes 0-15 row M=lane, K {0..15}; lanes 16-31 K {16..31}
__device__ __forceinline__ v16bf load_frag_a(const unsigned short* base, int stride, int lane) {
  const int r  = lane & 15;
  const int hh = lane >> 4;
  const unsigned short* p = base + r * stride + hh * 16;
  FragU f;
  f.u[0] = *(const uint4*)(p);
  f.u[1] = *(const uint4*)(p + 8);
  return f.v;
}

// B-fragment (K=32 x 16, bf16) from B^T [N][K] storage: lane n holds col n, K {0..15} / {16..31} per half
__device__ __forceinline__ v16bf load_frag_b(const unsigned short* base, int stride, int lane) {
  const int n  = lane & 15;
  const int hh = lane >> 4;
  const unsigned short* p = base + n * stride + hh * 16;
  FragU f;
  f.u[0] = *(const uint4*)(p);
  f.u[1] = *(const uint4*)(p + 8);
  return f.v;
}

__device__ __forceinline__ v8f wmma_bf16(v16bf a, v16bf b, v8f c) {
  return __builtin_amdgcn_wmma_f32_16x16x32_bf16(false, a, false, b, (short)0, c, false, false);
}

// ---------- fp32 -> bf16 bits elementwise ----------
__global__ __launch_bounds__(256) void cvt_kernel(const float* __restrict__ in,
                                                  unsigned short* __restrict__ out, int n) {
  int i = (blockIdx.x * 256 + threadIdx.x) * 4;
  if (i < n) {
    float4 v = *(const float4*)(in + i);
    union { unsigned short s[4]; uint2 u; } w;
    w.s[0] = f2bf_bits(v.x); w.s[1] = f2bf_bits(v.y);
    w.s[2] = f2bf_bits(v.z); w.s[3] = f2bf_bits(v.w);
    *(uint2*)(out + i) = w.u;
  }
}

// ---------- fp32 [K][N] -> bf16 bits transposed [N][K] (for weights) ----------
__global__ __launch_bounds__(256) void tcvt_kernel(const float* __restrict__ W,
                                                   unsigned short* __restrict__ Wt) {
  __shared__ float t[16][17];
  const int kt = blockIdx.x * 16, nt = blockIdx.y * 16;
  const int tx = threadIdx.x & 15, ty = threadIdx.x >> 4;
  t[ty][tx] = W[(size_t)(kt + ty) * D_ + nt + tx];
  __syncthreads();
  Wt[(size_t)(nt + ty) * D_ + kt + tx] = f2bf_bits(t[tx][ty]);
}

// ---------- tiled WMMA GEMM: C = A[4096x1024] * B^T-layout[1024x1024] (+bias, mode epilogue) ----------
// mode 0: Q proj -> [B,H,S,HD] bf16, scaled 1/sqrt(HD)
// mode 1: K proj -> [B,H,S,HD] bf16
// mode 2: V proj -> [B,H,HD,S] bf16 (transposed)
// mode 3: O proj -> fp32 [M,N] += bias + residual
__global__ __launch_bounds__(256) void gemm_kernel(const unsigned short* __restrict__ A,
                                                   const unsigned short* __restrict__ Bt,
                                                   const float* __restrict__ bias,
                                                   const float* __restrict__ resid,
                                                   void* __restrict__ outp, int mode) {
  constexpr int MK = D_;     // K dimension (1024)
  constexpr int LS = 72;     // LDS row stride: 64 + 8 pad (keeps rows 16B aligned)
  __shared__ __align__(16) unsigned short lA[2][128 * LS];   // double-buffered 128x64 tiles
  __shared__ __align__(16) unsigned short lB[2][128 * LS];
  const int tid = threadIdx.x;
  const int lane = tid & 31;
  const int wave = tid >> 5;
  const int waveM = wave >> 2;   // 0..1
  const int waveN = wave & 3;    // 0..3
  const int mBase = blockIdx.x * 128;
  const int nBase = blockIdx.y * 128;

  // staging map: thread -> rows {r0, r0+32, r0+64, r0+96}, 8-elem column c0 (16B units)
  const int r0 = tid >> 3;          // 0..31
  const int c0 = (tid & 7) * 8;     // 0..56
  const unsigned short* gA = A  + (size_t)(mBase + r0) * MK + c0;
  const unsigned short* gB = Bt + (size_t)(nBase + r0) * MK + c0;
  const int lo = r0 * LS + c0;
  unsigned sA[2][4], sB[2][4];      // LDS byte addresses (low 32 bits of generic)
  for (int p = 0; p < 2; ++p)
    for (int q = 0; q < 4; ++q) {
      sA[p][q] = (unsigned)(size_t)&lA[p][lo + q * 32 * LS];
      sB[p][q] = (unsigned)(size_t)&lB[p][lo + q * 32 * LS];
    }

  v8f acc[4][2];
  for (int i = 0; i < 4; ++i)
    for (int j = 0; j < 2; ++j)
      acc[i][j] = v8f{0.f, 0.f, 0.f, 0.f, 0.f, 0.f, 0.f, 0.f};

  auto issue_tile = [&](int p, int k0) {
    const unsigned short* a = gA + k0;
    const unsigned short* b = gB + k0;
    for (int q = 0; q < 4; ++q) {
      ASYNC_G2L_B128(sA[p][q], a + (size_t)(q * 32) * MK);
      ASYNC_G2L_B128(sB[p][q], b + (size_t)(q * 32) * MK);
    }
  };

  issue_tile(0, 0);   // prologue: stream tile 0 into buffer 0

  int buf = 0;
  for (int k0 = 0; k0 < MK; k0 += 64, buf ^= 1) {
    if (k0 + 64 < MK) {
      issue_tile(buf ^ 1, k0 + 64);
      asm volatile("s_wait_asynccnt 0x8" ::: "memory");   // current tile's 8 done, next 8 in flight
    } else {
      asm volatile("s_wait_asynccnt 0x0" ::: "memory");
    }
    __syncthreads();

    for (int kk = 0; kk < 64; kk += 32) {   // 16 WMMAs per barrier interval
      v16bf bfrag[2];
      for (int nf = 0; nf < 2; ++nf)
        bfrag[nf] = load_frag_b(&lB[buf][(waveN * 32 + nf * 16) * LS + kk], LS, lane);
      for (int mf = 0; mf < 4; ++mf) {
        v16bf afrag = load_frag_a(&lA[buf][(waveM * 64 + mf * 16) * LS + kk], LS, lane);
        for (int nf = 0; nf < 2; ++nf)
          acc[mf][nf] = wmma_bf16(afrag, bfrag[nf], acc[mf][nf]);
      }
    }
    __syncthreads();
  }

  const int col16 = lane & 15, hh = lane >> 4;
  for (int mf = 0; mf < 4; ++mf) {
    const int growB = mBase + waveM * 64 + mf * 16 + hh * 8;
    for (int nf = 0; nf < 2; ++nf) {
      const int gcol = nBase + waveN * 32 + nf * 16 + col16;
      const float bvl = bias[gcol];
      v8f a = acc[mf][nf];
      for (int r = 0; r < 8; ++r) {
        const int grow = growB + r;
        float val = a[r] + bvl;
        if (mode == 3) {
          ((float*)outp)[(size_t)grow * D_ + gcol] = val + resid[(size_t)grow * D_ + gcol];
        } else {
          const int b = grow >> 11, s = grow & (S_ - 1);
          const int h = gcol >> 6, hd = gcol & 63;
          unsigned short bf = f2bf_bits(mode == 0 ? val * 0.125f : val);
          unsigned short* o = (unsigned short*)outp;
          if (mode == 2)
            o[(((size_t)b * H_ + h) * HD_ + hd) * S_ + s] = bf;    // V^T layout
          else
            o[(((size_t)b * H_ + h) * S_ + s) * HD_ + hd] = bf;    // head layout
        }
      }
    }
  }
}

// ---------- flash attention: per wave 16 queries; S^T = K*Q^T, online softmax, O^T = V^T*P^T ----------
__global__ __launch_bounds__(256) void attn_kernel(const unsigned short* __restrict__ Qp,
                                                   const unsigned short* __restrict__ Kp,
                                                   const unsigned short* __restrict__ Vp,
                                                   const unsigned char* __restrict__ mask,
                                                   unsigned short* __restrict__ AO) {
  __shared__ __align__(16) unsigned short Plds[8][16][40];   // per-wave 16 queries x 32 keys (+pad)
  const int tid = threadIdx.x, lane = tid & 31, wave = tid >> 5;
  const int q16 = lane & 15, hh = lane >> 4;
  const int h = blockIdx.y, b = blockIdx.z;
  const int q0 = blockIdx.x * 128 + wave * 16;
  const size_t bh = (size_t)b * H_ + h;
  const unsigned short* Qb = Qp + bh * (size_t)S_ * HD_;
  const unsigned short* Kb = Kp + bh * (size_t)S_ * HD_;
  const unsigned short* Vb = Vp + bh * (size_t)HD_ * S_;
  const unsigned char* Mrow = mask + ((size_t)b * S_ + (q0 + q16)) * S_;

  const v16bf qb0 = load_frag_b(Qb + (size_t)q0 * HD_,      HD_, lane);  // hd 0..31
  const v16bf qb1 = load_frag_b(Qb + (size_t)q0 * HD_ + 32, HD_, lane);  // hd 32..63

  const v8f vz = v8f{0.f, 0.f, 0.f, 0.f, 0.f, 0.f, 0.f, 0.f};
  v8f o[4] = {vz, vz, vz, vz};           // O^T: hd rows (4x16), query cols
  float m = -1e30f, l = 0.f;

  for (int k0 = 0; k0 < S_; k0 += 32) {
    // issue all K fragments + mask bytes first (8 b128 loads in flight before first WMMA)
    const v16bf ka00 = load_frag_a(Kb + (size_t)k0 * HD_,             HD_, lane);
    const v16bf ka01 = load_frag_a(Kb + (size_t)k0 * HD_ + 32,       HD_, lane);
    const v16bf ka10 = load_frag_a(Kb + (size_t)(k0 + 16) * HD_,      HD_, lane);
    const v16bf ka11 = load_frag_a(Kb + (size_t)(k0 + 16) * HD_ + 32, HD_, lane);
    const unsigned long long m0 = *(const unsigned long long*)(Mrow + k0 + hh * 8);
    const unsigned long long m1 = *(const unsigned long long*)(Mrow + k0 + 16 + hh * 8);

    v8f st0 = wmma_bf16(ka00, qb0, vz);
    st0     = wmma_bf16(ka01, qb1, st0);
    v8f st1 = wmma_bf16(ka10, qb0, vz);
    st1     = wmma_bf16(ka11, qb1, st1);

    float s0[8], s1[8];
    float cmax = -1e30f;
    for (int r = 0; r < 8; ++r) {
      float x0 = ((m0 >> (8 * r)) & 0xffull) ? -1e30f : st0[r];
      float x1 = ((m1 >> (8 * r)) & 0xffull) ? -1e30f : st1[r];
      s0[r] = x0; s1[r] = x1;
      cmax = fmaxf(cmax, fmaxf(x0, x1));
    }
    cmax = fmaxf(cmax, __shfl_xor(cmax, 16, 32));    // merge lane-halves -> per-query max
    const float mnew = fmaxf(m, cmax);
    const float alpha = __expf(m - mnew);
    float lsum = 0.f;
    PackU w0, w1;
    for (int r = 0; r < 8; ++r) {
      float p0 = __expf(s0[r] - mnew);
      float p1 = __expf(s1[r] - mnew);
      lsum += p0 + p1;
      w0.s[r] = f2bf_bits(p0);
      w1.s[r] = f2bf_bits(p1);
    }
    lsum += __shfl_xor(lsum, 16, 32);
    l = l * alpha + lsum;
    m = mnew;
    for (int f = 0; f < 4; ++f) o[f] *= alpha;

    // stage P^T probs through wave-private LDS, read back as B fragment (lane=query, 16 keys/half)
    *(uint4*)&Plds[wave][q16][hh * 8]      = w0.u;
    *(uint4*)&Plds[wave][q16][16 + hh * 8] = w1.u;
    const v16bf pfrag = load_frag_b(&Plds[wave][0][0], 40, lane);

    // all V^T fragments in flight before the PV WMMAs
    v16bf va0 = load_frag_a(Vb + (size_t)(0 * 16) * S_ + k0, S_, lane);
    v16bf va1 = load_frag_a(Vb + (size_t)(1 * 16) * S_ + k0, S_, lane);
    v16bf va2 = load_frag_a(Vb + (size_t)(2 * 16) * S_ + k0, S_, lane);
    v16bf va3 = load_frag_a(Vb + (size_t)(3 * 16) * S_ + k0, S_, lane);
    o[0] = wmma_bf16(va0, pfrag, o[0]);
    o[1] = wmma_bf16(va1, pfrag, o[1]);
    o[2] = wmma_bf16(va2, pfrag, o[2]);
    o[3] = wmma_bf16(va3, pfrag, o[3]);
  }

  const float inv = 1.f / l;
  const int s = q0 + q16;
  for (int f = 0; f < 4; ++f)
    for (int r = 0; r < 8; ++r) {
      const int hd = f * 16 + hh * 8 + r;
      AO[((size_t)b * S_ + s) * D_ + h * HD_ + hd] = f2bf_bits(o[f][r] * inv);
    }
}

// ---------- residual already added; LayerNorm (unbiased std, (std+eps) denom) ----------
__global__ __launch_bounds__(256) void ln_kernel(const float* __restrict__ Y,
                                                 const float* __restrict__ gamma,
                                                 const float* __restrict__ beta,
                                                 float* __restrict__ out) {
  __shared__ float red[2][8];
  const int row = blockIdx.x, tid = threadIdx.x;
  const float* x = Y + (size_t)row * D_;
  float v[4], sum = 0.f, sq = 0.f;
  for (int k = 0; k < 4; ++k) {
    v[k] = x[tid + k * 256];
    sum += v[k]; sq += v[k] * v[k];
  }
  for (int off = 16; off > 0; off >>= 1) {
    sum += __shfl_xor(sum, off, 32);
    sq  += __shfl_xor(sq,  off, 32);
  }
  const int wave = tid >> 5, lane = tid & 31;
  if (lane == 0) { red[0][wave] = sum; red[1][wave] = sq; }
  __syncthreads();
  if (tid == 0) {
    float s = 0.f, q = 0.f;
    for (int i = 0; i < 8; ++i) { s += red[0][i]; q += red[1][i]; }
    red[0][0] = s; red[1][0] = q;
  }
  __syncthreads();
  sum = red[0][0]; sq = red[1][0];
  const float mean = sum / (float)D_;
  const float var  = (sq - (float)D_ * mean * mean) / (float)(D_ - 1);
  const float inv  = 1.f / (sqrtf(fmaxf(var, 0.f)) + EPS_);
  for (int k = 0; k < 4; ++k) {
    const int i = tid + k * 256;
    out[(size_t)row * D_ + i] = gamma[i] * (v[k] - mean) * inv + beta[i];
  }
}

extern "C" void kernel_launch(void* const* d_in, const int* in_sizes, int n_in,
                              void* d_out, int out_size, void* d_ws, size_t ws_size,
                              hipStream_t stream) {
  (void)in_sizes; (void)n_in; (void)out_size; (void)ws_size;
  const float* query = (const float*)d_in[0];
  const float* key   = (const float*)d_in[1];
  const float* value = (const float*)d_in[2];
  const unsigned char* mask = (const unsigned char*)d_in[3];
  const float* Wq = (const float*)d_in[4];
  const float* bq = (const float*)d_in[5];
  const float* Wk = (const float*)d_in[6];
  const float* bk = (const float*)d_in[7];
  const float* Wv = (const float*)d_in[8];
  const float* bv = (const float*)d_in[9];
  const float* Wo = (const float*)d_in[10];
  const float* bo = (const float*)d_in[11];
  const float* gamma = (const float*)d_in[12];
  const float* beta  = (const float*)d_in[13];
  float* out = (float*)d_out;

  const size_t NEL = (size_t)B_ * S_ * D_;   // 4 Mi elems
  const size_t WEL = (size_t)D_ * D_;        // 1 Mi elems
  char* w = (char*)d_ws;
  auto carve = [&](size_t bytes) -> void* {
    void* p = (void*)w;
    w += (bytes + 255) & ~(size_t)255;
    return p;
  };
  unsigned short* Xq  = (unsigned short*)carve(NEL * 2);
  unsigned short* Xk  = (unsigned short*)carve(NEL * 2);
  unsigned short* Xv  = (unsigned short*)carve(NEL * 2);
  unsigned short* Wqt = (unsigned short*)carve(WEL * 2);
  unsigned short* Wkt = (unsigned short*)carve(WEL * 2);
  unsigned short* Wvt = (unsigned short*)carve(WEL * 2);
  unsigned short* Wot = (unsigned short*)carve(WEL * 2);
  unsigned short* Qp  = (unsigned short*)carve(NEL * 2);
  unsigned short* Kp  = (unsigned short*)carve(NEL * 2);
  unsigned short* Vp  = (unsigned short*)carve(NEL * 2);
  unsigned short* AO  = (unsigned short*)carve(NEL * 2);
  float* Yt = (float*)Xq;   // reuse Xq+Xk region (16 MB) after projections are done

  const dim3 blk(256);
  cvt_kernel<<<dim3((unsigned)(NEL / 1024)), blk, 0, stream>>>(query, Xq, (int)NEL);
  cvt_kernel<<<dim3((unsigned)(NEL / 1024)), blk, 0, stream>>>(key,   Xk, (int)NEL);
  cvt_kernel<<<dim3((unsigned)(NEL / 1024)), blk, 0, stream>>>(value, Xv, (int)NEL);
  const dim3 tg(D_ / 16, D_ / 16);
  tcvt_kernel<<<tg, blk, 0, stream>>>(Wq, Wqt);
  tcvt_kernel<<<tg, blk, 0, stream>>>(Wk, Wkt);
  tcvt_kernel<<<tg, blk, 0, stream>>>(Wv, Wvt);
  tcvt_kernel<<<tg, blk, 0, stream>>>(Wo, Wot);

  const int M = B_ * S_;                        // 4096
  const dim3 gg(M / 128, D_ / 128);             // 32 x 8
  gemm_kernel<<<gg, blk, 0, stream>>>(Xq, Wqt, bq, nullptr, (void*)Qp, 0);
  gemm_kernel<<<gg, blk, 0, stream>>>(Xk, Wkt, bk, nullptr, (void*)Kp, 1);
  gemm_kernel<<<gg, blk, 0, stream>>>(Xv, Wvt, bv, nullptr, (void*)Vp, 2);

  attn_kernel<<<dim3(S_ / 128, H_, B_), blk, 0, stream>>>(Qp, Kp, Vp, mask, AO);

  gemm_kernel<<<gg, blk, 0, stream>>>(AO, Wot, bo, query, (void*)Yt, 3);
  ln_kernel<<<dim3(M), blk, 0, stream>>>(Yt, gamma, beta, out);
}
